// AttentionDownSample_15899968930174
// MI455X (gfx1250) — compile-verified, matrix operand order checked by reading, not run on hardware
//
#include <hip/hip_runtime.h>

typedef __attribute__((ext_vector_type(16))) __bf16    v16bf;
typedef __attribute__((ext_vector_type(8)))  __bf16    v8bf;
typedef __attribute__((ext_vector_type(8)))  float     v8f;
typedef __attribute__((ext_vector_type(4)))  unsigned  v4u;
typedef __attribute__((ext_vector_type(8)))  unsigned  v8u;

#define C_DIM   128   // channels (GEMM K)
#define R_DIM   32    // reduce dim (GEMM M)
#define H_IN    256
#define W_IN    256
#define H_OUT   128
#define W_OUT   128
#define TILE_J  32            // coarse pixels per block
#define COLS    (TILE_J * 4)  // subpixel columns = 128
#define NTHREADS 128

__global__ __launch_bounds__(NTHREADS)
void attn_downsample_kernel(const float* __restrict__ fm,
                            const float* __restrict__ Wq,
                            const float* __restrict__ Wk,
                            float* __restrict__ out)
{
    // LDS: 64K + 32K + 8K + 16K + 4K + 16K + 0.5K ~= 140.5 KB (< 320 KB/WGP)
    __shared__ float   s_raw[C_DIM * COLS];      // TDM dest: [c][dy][x] = c*128 + dy*64 + x
    __shared__ __bf16  s_fmbf[COLS * C_DIM];     // [col][c] bf16 (K-contiguous WMMA B), col = p*4 + dy*2 + dx
    __shared__ __bf16  s_pool[TILE_J * C_DIM];   // [p][c]  bf16 pooled
    __shared__ __bf16  s_w[2 * R_DIM * C_DIM];   // [r][c]  bf16: rows 0-31 Wq, 32-63 Wk
    __shared__ float   s_q[R_DIM * TILE_J];      // [r][p]
    __shared__ float   s_k[R_DIM * COLS];        // [r][col]
    __shared__ float   s_attn[COLS];

    const int t    = threadIdx.x;
    const int lane = t & 31;
    const int wave = t >> 5;
    const int jt = blockIdx.x;   // 0..3   (32 coarse cols each)
    const int i  = blockIdx.y;   // 0..127 coarse row
    const int b  = blockIdx.z;   // 0..7
    const int x0 = jt * 64;

    const float* fmb = fm + (size_t)b * C_DIM * H_IN * W_IN;

    // ---------------- stage 1a: TDM DMA of the fm tile into LDS (wave 0 issues) ----
    // Tile: dim0 = x (64 contig floats), dim1 = dy (2 rows, stride 256),
    //       dim2 = c (128 planes, stride 65536).  64 KB total, packed linearly.
    if (wave == 0) {
        const unsigned long long ga =
            (unsigned long long)(uintptr_t)(fmb + (size_t)(2 * i) * W_IN + x0);
        const unsigned lds = (unsigned)(uintptr_t)(&s_raw[0]);  // low 32 bits = LDS offset

        // D# group 0: count=1 | lds_addr[63:32] | global_addr[120:64] | type=2
        v4u g0 = { 1u,
                   lds,
                   (unsigned)(ga & 0xFFFFFFFFu),
                   (unsigned)((ga >> 32) & 0x01FFFFFFu) | (2u << 30) };
        // D# group 1
        const unsigned td0 = W_IN, td1 = H_IN;          // tensor dims (no OOB hit)
        const unsigned st0 = W_IN;                      // dim0 stride (elems)
        const unsigned st1 = H_IN * W_IN;               // dim1 stride (elems)
        v8u g1 = { (2u << 16),                          // wg_mask=0 | data_size=4B | no flags
                   (td0 & 0xFFFFu) << 16,               // barrier_addr=0 | tensor_dim0.lo
                   (td0 >> 16) | ((td1 & 0xFFFFu) << 16),
                   (td1 >> 16) | (64u << 16),           // tile_dim0 = 64
                   2u | (128u << 16),                   // tile_dim1 = 2, tile_dim2 = 128
                   st0,                                 // tensor_dim0_stride lo32
                   0u | ((st1 & 0xFFFFu) << 16),        // st0 hi16 | st1 lo16
                   (st1 >> 16) };                       // st1 hi32
        // D# group 2: tensor_dim2=128, tensor_dim3=0, dim2_stride=0 (tile_dim3=0)
        v4u g2 = { 128u, 0u, 0u, 0u };
        v4u g3 = { 0u, 0u, 0u, 0u };

        asm volatile("tensor_load_to_lds %0, %1, %2, %3"
                     :: "s"(g0), "s"(g1), "s"(g2), "s"(g3)
                     : "memory");
    }

    // ---------------- stage 1b: weights -> LDS bf16 (all waves, overlaps DMA) ------
    #pragma unroll 4
    for (int k = 0; k < 64; ++k) {
        int idx = k * NTHREADS + t;   // 0..8191
        float wv = (idx < R_DIM * C_DIM) ? Wq[idx] : Wk[idx - R_DIM * C_DIM];
        s_w[idx] = (__bf16)wv;
    }
    if (wave == 0) {
        __builtin_amdgcn_s_wait_tensorcnt(0);   // retire the TDM transfer
    }
    __syncthreads();

    // ---------------- stage 2: fused bf16 repack + 2x2 average pool ----------------
    // thread t = channel c; raw [c][dy][x] -> bf16 [col][c] and pooled [p][c]
    #pragma unroll 4
    for (int x4 = 0; x4 < 16; ++x4) {
        const float4 v0 = *(const float4*)&s_raw[t * COLS + x4 * 4];        // dy=0
        const float4 v1 = *(const float4*)&s_raw[t * COLS + 64 + x4 * 4];   // dy=1
        const int p0 = 2 * x4, p1 = 2 * x4 + 1;
        s_fmbf[(p0 * 4 + 0) * C_DIM + t] = (__bf16)v0.x;
        s_fmbf[(p0 * 4 + 1) * C_DIM + t] = (__bf16)v0.y;
        s_fmbf[(p0 * 4 + 2) * C_DIM + t] = (__bf16)v1.x;
        s_fmbf[(p0 * 4 + 3) * C_DIM + t] = (__bf16)v1.y;
        s_fmbf[(p1 * 4 + 0) * C_DIM + t] = (__bf16)v0.z;
        s_fmbf[(p1 * 4 + 1) * C_DIM + t] = (__bf16)v0.w;
        s_fmbf[(p1 * 4 + 2) * C_DIM + t] = (__bf16)v1.z;
        s_fmbf[(p1 * 4 + 3) * C_DIM + t] = (__bf16)v1.w;
        s_pool[p0 * C_DIM + t] = (__bf16)((v0.x + v0.y + v1.x + v1.y) * 0.25f);
        s_pool[p1 * C_DIM + t] = (__bf16)((v0.z + v0.w + v1.z + v1.w) * 0.25f);
    }
    __syncthreads();

    // ---------------- stage 3: WMMA GEMMs (bf16 -> f32) ----------------
    // tiles 0..3  : Q = Wq(32x128) @ pooled(128x32)   -> s_q
    // tiles 4..19 : K = Wk(32x128) @ fm_sub(128x128)  -> s_k
    for (int tile = wave; tile < 20; tile += 4) {
        const bool isQ = (tile < 4);
        int mt, nt, ostride;
        const __bf16* Bbase;
        float* Obase;
        if (isQ) { mt = tile >> 1;   nt = tile & 1;  Bbase = s_pool; Obase = s_q; ostride = TILE_J; }
        else     { int u = tile - 4; mt = u >> 3; nt = u & 7; Bbase = s_fmbf; Obase = s_k; ostride = COLS; }
        const __bf16* Abase = s_w + (isQ ? 0 : R_DIM * C_DIM);

        const int arow  = mt * 16 + (lane & 15);
        const int bcol  = nt * 16 + (lane & 15);
        const int khalf = (lane < 16) ? 0 : 8;   // 16-bit A/B lane K-split

        v8f acc = {};
        #pragma unroll
        for (int ch = 0; ch < 4; ++ch) {         // K = 128 in chunks of 32
            int kb = ch * 32 + khalf;
            v8bf a0 = *(const v8bf*)&Abase[arow * C_DIM + kb];
            v8bf a1 = *(const v8bf*)&Abase[arow * C_DIM + kb + 16];
            v8bf b0 = *(const v8bf*)&Bbase[bcol * C_DIM + kb];
            v8bf b1 = *(const v8bf*)&Bbase[bcol * C_DIM + kb + 16];
            v16bf A, B;
            #pragma unroll
            for (int e = 0; e < 8; ++e) { A[e] = a0[e]; A[e + 8] = a1[e];
                                          B[e] = b0[e]; B[e + 8] = b1[e]; }
            acc = __builtin_amdgcn_wmma_f32_16x16x32_bf16(
                false, A, false, B, (short)0, acc, false, false);
        }
        // C layout: VGPR v -> M = v + 8*(lane>=16), N = lane%16
        const int rbase = mt * 16 + ((lane >> 4) << 3);
        const int cc    = nt * 16 + (lane & 15);
        #pragma unroll
        for (int v = 0; v < 8; ++v)
            Obase[(rbase + v) * ostride + cc] = acc[v];
    }
    __syncthreads();

    // ---------------- stage 4: logits + softmax over 4-wide window ----------------
    {
        const int p = t >> 2, s = t & 3;
        float l = 0.f;
        #pragma unroll 8
        for (int r = 0; r < R_DIM; ++r)
            l += s_q[r * TILE_J + p] * s_k[r * COLS + p * 4 + s];
        l *= 0.17677669529663687f;   // 1/sqrt(32) applied to Q branch
        float m = l;
        m = fmaxf(m, __shfl_xor(m, 1, 32));
        m = fmaxf(m, __shfl_xor(m, 2, 32));
        float e = __expf(l - m);
        float d = e;
        d += __shfl_xor(d, 1, 32);
        d += __shfl_xor(d, 2, 32);
        s_attn[t] = e / d;
    }
    __syncthreads();

    // ---------------- stage 5: V weighted sum (fp32), coalesced stores ----------------
    float* outb = out + (size_t)b * C_DIM * H_OUT * W_OUT;
    #pragma unroll 4
    for (int k = 0; k < 32; ++k) {
        int c = k * 4 + (t >> 5);
        int p = t & 31;
        const float2 g0 = *(const float2*)&s_raw[c * COLS + p * 2];        // dy=0: s=0,1
        const float2 g1 = *(const float2*)&s_raw[c * COLS + 64 + p * 2];   // dy=1: s=2,3
        const float4 a  = *(const float4*)&s_attn[p * 4];
        float o = g0.x * a.x + g0.y * a.y + g1.x * a.z + g1.y * a.w;
        outb[((size_t)c * H_OUT + i) * W_OUT + jt * TILE_J + p] = o;
    }
}

extern "C" void kernel_launch(void* const* d_in, const int* in_sizes, int n_in,
                              void* d_out, int out_size, void* d_ws, size_t ws_size,
                              hipStream_t stream) {
    const float* fm = (const float*)d_in[0];
    const float* Wq = (const float*)d_in[1];
    const float* Wk = (const float*)d_in[2];
    float* out = (float*)d_out;
    (void)in_sizes; (void)n_in; (void)out_size; (void)d_ws; (void)ws_size;
    dim3 grid(W_OUT / TILE_J, H_OUT, 8);
    dim3 block(NTHREADS);
    attn_downsample_kernel<<<grid, block, 0, stream>>>(fm, Wq, Wk, out);
}
